// CrossAttention_76098230550569
// MI455X (gfx1250) — matrix operand order repaired and checked by reading, not compiled, not run on hardware
//
#include <hip/hip_runtime.h>
#include <cmath>

typedef __attribute__((ext_vector_type(16))) _Float16 v16h;
typedef __attribute__((ext_vector_type(8)))  _Float16 v8h;
typedef __attribute__((ext_vector_type(8)))  float    v8f;
typedef __attribute__((ext_vector_type(4)))  float    v4f;

#define NBATCH 16
#define LQ 2048
#define LK 2048
#define DD 256
#define DVV 256
#define SCALE 0.0625f   // 1/sqrt(256)
#define WAVES 8
#define TK 32           // keys per chunk

// padded LDS strides (in halves); all multiples of 16/8 keep vector alignment
#define KSTRIDE 272     // K tile row stride (256 feats + 16 pad)
#define VSTRIDE 48      // V^T tile row stride (32 keys + 16 pad)
#define PSTRIDE 40      // P stage row stride (32 cols + 8 pad)

#define KT_HALVES   (TK * KSTRIDE)            // 8704  per buffer
#define VT_HALVES   (DVV * VSTRIDE)           // 12288 per buffer
#define PS_HALVES   (16 * PSTRIDE)            // 640   per wave
#define VT_BASE     (2 * KT_HALVES)
#define PS_BASE     (2 * KT_HALVES + 2 * VT_HALVES)
#define SMEM_HALVES (PS_BASE + WAVES * PS_HALVES)
#define SMEM_BYTES  (SMEM_HALVES * 2)         // 94208 bytes

// Cooperative stage of one 32-key chunk into LDS. Integer offsets off the
// extern-shared array only (keeps everything in addrspace(3) -> ds_* ops).
__device__ __forceinline__ void stage_tiles(const float* __restrict__ k,
                                            const float* __restrict__ v,
                                            int k0, unsigned ktoff, unsigned vtoff,
                                            int tid)
{
    extern __shared__ _Float16 smem[];
    // ---- K tile row-major: thread -> (key = tid>>3, feats f0..f0+31) ----
    {
        const int key = tid >> 3;
        const int f0  = (tid & 7) * 32;
        const float* src = k + (size_t)(k0 + key) * DD + f0;
        const unsigned d = ktoff + key * KSTRIDE + f0;
        #pragma unroll
        for (int g = 0; g < 2; ++g) {
            v16h tmp;
            #pragma unroll
            for (int j = 0; j < 16; ++j) tmp[j] = (_Float16)src[g * 16 + j];
            *(v16h*)&smem[d + g * 16] = tmp;
        }
    }
    // ---- V tile transposed [col][key] ----
    // thread -> 4 columns x 8 keys patch: coalesced float4 row loads,
    // 4 contiguous v8h transposed stores.
    {
        const int col0 = (tid & 63) * 4;      // 0..252
        const int kb   = (tid >> 6) * 8;      // 0,8,16,24
        v4f rowv[8];
        const float* src = v + (size_t)(k0 + kb) * DVV + col0;
        #pragma unroll
        for (int j = 0; j < 8; ++j)
            rowv[j] = *(const v4f*)(src + (size_t)j * DVV);
        #pragma unroll
        for (int c = 0; c < 4; ++c) {
            v8h tmp;
            #pragma unroll
            for (int j = 0; j < 8; ++j) tmp[j] = (_Float16)rowv[j][c];
            *(v8h*)&smem[vtoff + (unsigned)(col0 + c) * VSTRIDE + kb] = tmp;
        }
    }
}

__global__ __launch_bounds__(256, 1)
void cross_attn_fa_kernel(const float* __restrict__ Q, const float* __restrict__ K,
                          const float* __restrict__ V, const int* __restrict__ VL,
                          float* __restrict__ O)
{
    extern __shared__ _Float16 smem[];

    const int tid   = threadIdx.x;
    const int lane  = tid & 31;
    const int wave  = tid >> 5;
    const int batch = blockIdx.y;
    const int qrow0 = blockIdx.x * (WAVES * 16) + wave * 16;

    const float* q = Q + (size_t)batch * LQ * DD;
    const float* k = K + (size_t)batch * LK * DD;
    const float* v = V + (size_t)batch * LK * DVV;
    float*       o = O + (size_t)batch * LQ * DVV;
    const int vlen = VL[batch];

    const int nlane = lane & 15;  // N (col) in B/C/D layout; M (row) in A layout
    const int hi    = lane >> 4;  // half-wave select
    const unsigned psoff = PS_BASE + (unsigned)wave * PS_HALVES;  // wave-private

    // ---- load Q strip (16 x 256) into A-layout f16 registers (once) ----
    v16h qa[8];
    {
        const int koff = hi * 16;
        for (int c = 0; c < 8; ++c) {
            const float* qr = q + (size_t)(qrow0 + nlane) * DD + c * 32;
            #pragma unroll
            for (int j = 0; j < 16; ++j) {
                const int kk = koff + j;
                qa[c][j] = (_Float16)qr[kk];
            }
        }
    }

    v8f acc[16];
    #pragma unroll
    for (int t = 0; t < 16; ++t) acc[t] = (v8f)0.0f;

    float mrow[8], lrow[8];
    #pragma unroll
    for (int r = 0; r < 8; ++r) { mrow[r] = -INFINITY; lrow[r] = 0.0f; }

    constexpr int NCHUNK = LK / TK;
    stage_tiles(k, v, 0, 0u, VT_BASE, tid);

    for (int i = 0; i < NCHUNK; ++i) {
        const int k0  = i * TK;
        const unsigned buf = (unsigned)(i & 1);
        __syncthreads();   // chunk i staged; other buffer free for reuse
        if (i + 1 < NCHUNK)
            stage_tiles(k, v, k0 + TK,
                        (buf ^ 1u) * KT_HALVES,
                        VT_BASE + (buf ^ 1u) * VT_HALVES, tid);

        const unsigned ktb = buf * KT_HALVES;              // integer offsets only
        const unsigned vtb = VT_BASE + buf * VT_HALVES;

        // ----- S = (Q K^T), two independent 16x16 chains interleaved -----
        v8f s0 = (v8f)0.0f, s1 = (v8f)0.0f;
        {
            const unsigned a0 = ktb + (unsigned)nlane * KSTRIDE + hi * 16;
            const unsigned a1 = a0 + 16u * KSTRIDE;
            #pragma unroll
            for (int c = 0; c < 8; ++c) {
                v16h b0 = *(const v16h*)&smem[a0 + c * 32];
                v16h b1 = *(const v16h*)&smem[a1 + c * 32];
                s0 = __builtin_amdgcn_wmma_f32_16x16x32_f16(
                    false, qa[c], false, b0, (short)0, s0, false, false);
                s1 = __builtin_amdgcn_wmma_f32_16x16x32_f16(
                    false, qa[c], false, b1, (short)0, s1, false, false);
            }
        }
        // scale + SequenceMask: masked scores become 0 BEFORE softmax
        {
            const bool cv0 = (k0 + nlane) < vlen;
            const bool cv1 = (k0 + 16 + nlane) < vlen;
            #pragma unroll
            for (int r = 0; r < 8; ++r) {
                s0[r] = cv0 ? s0[r] * SCALE : 0.0f;
                s1[r] = cv1 ? s1[r] * SCALE : 0.0f;
            }
        }

        // ----- online softmax over the 32 new columns -----
        float p0[8], p1[8], cf[8];
        #pragma unroll
        for (int r = 0; r < 8; ++r) {
            float tmax = fmaxf(s0[r], s1[r]);
            #pragma unroll
            for (int off = 8; off >= 1; off >>= 1)
                tmax = fmaxf(tmax, __shfl_xor(tmax, off, 32));
            const float mnew = fmaxf(mrow[r], tmax);
            cf[r]   = __expf(mrow[r] - mnew);
            mrow[r] = mnew;
            p0[r] = __expf(s0[r] - mnew);
            p1[r] = __expf(s1[r] - mnew);
            float rs = p0[r] + p1[r];
            #pragma unroll
            for (int off = 8; off >= 1; off >>= 1)
                rs += __shfl_xor(rs, off, 32);
            lrow[r] = lrow[r] * cf[r] + rs;
        }

        // rescale running accumulators
        #pragma unroll
        for (int t = 0; t < 16; ++t) {
            #pragma unroll
            for (int r = 0; r < 8; ++r) acc[t][r] *= cf[r];
        }

        // ----- stage P (16x32 f16) in LDS (C-layout writes, wave-private) -----
        {
            const int rbase = hi * 8;
            #pragma unroll
            for (int r = 0; r < 8; ++r) {
                smem[psoff + (rbase + r) * PSTRIDE + nlane]      = (_Float16)p0[r];
                smem[psoff + (rbase + r) * PSTRIDE + 16 + nlane] = (_Float16)p1[r];
            }
        }
        // LDS ops from one wave stay in order; region is wave-private.

        // ----- read P back in A layout (two contiguous 8-half runs) -----
        v16h pa;
        {
            const int koff = hi * 16;
            v8h lo  = *(const v8h*)&smem[psoff + nlane * PSTRIDE + koff];
            v8h hi8 = *(const v8h*)&smem[psoff + nlane * PSTRIDE + 8 + koff];
            #pragma unroll
            for (int j = 0; j < 8; ++j) { pa[j] = lo[j]; pa[8 + j] = hi8[j]; }
        }

        // ----- O += P * V : batch loads, 4 independent WMMAs per group -----
        {
            const unsigned vbase = vtb + (unsigned)nlane * VSTRIDE + hi * 16;
            #pragma unroll
            for (int g = 0; g < 4; ++g) {
                v16h vb[4];
                #pragma unroll
                for (int u = 0; u < 4; ++u)
                    vb[u] = *(const v16h*)&smem[vbase + (g * 4 + u) * 16u * VSTRIDE];
                #pragma unroll
                for (int u = 0; u < 4; ++u)
                    acc[g * 4 + u] = __builtin_amdgcn_wmma_f32_16x16x32_f16(
                        false, pa, false, vb[u], (short)0, acc[g * 4 + u],
                        false, false);
            }
        }
    }

    // ----- normalize and store (C layout: row = rbase+r, col = t*16+nlane) -----
    const int rbase = hi * 8;
    #pragma unroll
    for (int t = 0; t < 16; ++t) {
        #pragma unroll
        for (int r = 0; r < 8; ++r) {
            const float val = acc[t][r] / lrow[r];
            o[(size_t)(qrow0 + rbase + r) * DVV + t * 16 + nlane] = val;
        }
    }
}

extern "C" void kernel_launch(void* const* d_in, const int* in_sizes, int n_in,
                              void* d_out, int out_size, void* d_ws, size_t ws_size,
                              hipStream_t stream) {
    const float* q  = (const float*)d_in[0];
    const float* k  = (const float*)d_in[1];
    const float* v  = (const float*)d_in[2];
    const int*   vl = (const int*)d_in[3];
    float*       o  = (float*)d_out;

    dim3 grid(LQ / (WAVES * 16), NBATCH);   // (16, 16)
    cross_attn_fa_kernel<<<grid, WAVES * 32, SMEM_BYTES, stream>>>(q, k, v, vl, o);
}
